// Decoder_7705171329099
// MI455X (gfx1250) — matrix operand order repaired and checked by reading, not compiled, not run on hardware
//
#include <hip/hip_runtime.h>
#include <hip/hip_bf16.h>

// ---------------------------------------------------------------------------
// Problem constants (match the reference)
// ---------------------------------------------------------------------------
#define GLOVE 300
#define H2    1024
#define H3    (3 * H2)     // 3072
#define VOCAB 32000
#define BSZ   32
#define TDEC  64
#define TENC  64
#define BT    (BSZ * TDEC) // 2048

// GEMM macro-tile (per 256-thread block): 32 (M) x 256 (N), K-step 32.
// 8 wave32s arranged 2 (M) x 4 (N); each wave computes 16x64 = 4 WMMA tiles.
#define BM 32
#define BN 256
#define BK 32
#define LPAD 4   // row stride 36 elems = 72 B -> 8 B-aligned rows for b64 LDS ops

typedef __attribute__((ext_vector_type(16))) __bf16 v16bf;
typedef __attribute__((ext_vector_type(4)))  __bf16 v4bf;
typedef __attribute__((ext_vector_type(8)))  float  v8f;

// ---------------------------------------------------------------------------
// Generic strided/batched WMMA GEMM:
//   C[b,m,n] = act( sum_k A[b,m,k]*B[b,k,n] + bias[n] + Cacc[b,m,n] )
// A element:  A + b*Abat + m*Ars + k*Acs
// B element:  B + b*Bbat + k*Brs + n*Bcs
// C element:  C + b*Cbat + m*Crs + n
// Requirements (all call sites satisfy): M % 32 == 0, N % 16 == 0,
// and for the vector fast paths: strides are multiples of 4 floats.
// fp32 in/out, bf16 WMMA compute with f32 accumulate.
// ---------------------------------------------------------------------------
__global__ __launch_bounds__(256) void wmma_gemm_kernel(
    const float* __restrict__ A, const float* __restrict__ B,
    const float* __restrict__ bias, const float* __restrict__ Cacc,
    float* __restrict__ C,
    int M, int N, int K,
    long long Ars, long long Acs, long long Brs, long long Bcs,
    long long Crs,
    long long Abat, long long Bbat, long long Cbat,
    int act)
{
    __shared__ __bf16 lA[BM][BK + LPAD];   // [m][k]
    __shared__ __bf16 lB[BN][BK + LPAD];   // [n][k]  (transposed stage)

    const int tid  = threadIdx.x;
    const int wave = tid >> 5;
    const int lane = tid & 31;
    const int half = lane >> 4;       // 0: lanes 0-15, 1: lanes 16-31
    const int l15  = lane & 15;
    const int wm   = wave >> 2;       // 0..1  (M sub-tile)
    const int wn   = wave & 3;        // 0..3  (N sub-tile group)

    const int n0 = blockIdx.x * BN;   // block N origin
    const int m0 = blockIdx.y * BM;   // block M origin (always valid: M%32==0)
    const int b  = blockIdx.z;

    const float* Ab = A + (long long)b * Abat;
    const float* Bb = B + (long long)b * Bbat;

    // cooperative-load coordinates (generic path)
    const int akk = tid & 31;         // A: k within tile
    const int am0 = tid >> 5;         // A: m stride-8 start
    // fast-path A coords: one float4 per thread
    const int afr = tid >> 3;         // A row 0..31
    const int afq = (tid & 7) << 2;   // A k-quad 0,4,...,28
    // B coords
    const int bn  = tid;              // B: n within tile (0..255)
    const int gn  = n0 + bn;
    const long long ncl = (gn < N) ? gn : (N - 1);   // clamped col (branch-free)

    v8f acc0 = {}, acc1 = {}, acc2 = {}, acc3 = {};

    const int kTiles = (K + BK - 1) / BK;
    for (int kt = 0; kt < kTiles; ++kt) {
        const int k0 = kt * BK;
        const bool fullK = (k0 + BK <= K);   // wave-uniform

        // speculative prefetch of next K-tile (global_prefetch_b8)
        if (kt + 1 < kTiles) {
            __builtin_prefetch(&Ab[(long long)(m0 + afr) * Ars + (long long)(k0 + BK) * Acs], 0, 1);
            __builtin_prefetch(&Bb[(long long)(k0 + BK) * Brs + ncl * Bcs], 0, 1);
        }

        // --- stage A tile: 32x32 fp32 -> bf16 ---
        if (Acs == 1 && fullK) {
            // vector fast path: one global_load_b128 per thread
            const float4 f = *(const float4*)&Ab[(long long)(m0 + afr) * Ars + (k0 + afq)];
            v4bf q;
            q[0] = (__bf16)f.x; q[1] = (__bf16)f.y;
            q[2] = (__bf16)f.z; q[3] = (__bf16)f.w;
            *(v4bf*)&lA[afr][afq] = q;
        } else {
            const int k = k0 + akk;
            const long long kcl = (k < K) ? k : (K - 1);
            const long long colOff = kcl * Acs;
            #pragma unroll
            for (int j = 0; j < 4; ++j) {
                const int m = am0 + 8 * j;
                float v = Ab[(long long)(m0 + m) * Ars + colOff];
                if (k >= K) v = 0.0f;
                lA[m][akk] = (__bf16)v;
            }
        }

        // --- stage B tile: 32x256 fp32 -> bf16 (transposed: lB[n][k]) ---
        if (Brs == 1 && fullK) {
            // vector fast path: 8 x global_load_b128 per thread (contiguous K)
            const float* src = &Bb[ncl * Bcs + k0];
            #pragma unroll
            for (int q = 0; q < 8; ++q) {
                const float4 f = *(const float4*)&src[4 * q];
                v4bf w;
                w[0] = (__bf16)f.x; w[1] = (__bf16)f.y;
                w[2] = (__bf16)f.z; w[3] = (__bf16)f.w;
                *(v4bf*)&lB[bn][4 * q] = w;
            }
        } else {
            const long long colOff = ncl * Bcs;
            #pragma unroll 8
            for (int kk = 0; kk < BK; ++kk) {
                const int k = k0 + kk;
                const long long kcl = (k < K) ? k : (K - 1);
                float v = Bb[kcl * Brs + colOff];
                if (k >= K || gn >= N) v = 0.0f;
                lB[bn][kk] = (__bf16)v;
            }
        }

        __syncthreads();

        // --- A fragment (16x32 bf16, ISA 7.12.2 layout), reused 4x ---
        v16bf af;
        {
            const int ar = wm * 16 + l15;
            #pragma unroll
            for (int i = 0; i < 16; ++i) {
                const int kpat = (i < 8) ? (half * 8 + i) : (16 + half * 8 + (i - 8));
                af[i] = lA[ar][kpat];
            }
        }

        // --- 4 B fragments + 4 WMMAs ---
        #pragma unroll
        for (int j = 0; j < 4; ++j) {
            const int bc = wn * 64 + j * 16 + l15;
            v16bf bfr;
            #pragma unroll
            for (int i = 0; i < 16; ++i)
                bfr[i] = lB[bc][(i < 8) ? (half * 8 + i) : (16 + half * 8 + (i - 8))];
            v8f* accp = (j == 0) ? &acc0 : (j == 1) ? &acc1 : (j == 2) ? &acc2 : &acc3;
            *accp = __builtin_amdgcn_wmma_f32_16x16x32_bf16(
                false, af, false, bfr, (short)0, *accp, false, false);
        }

        __syncthreads();
    }

    // --- epilogue: per-tile uniform N guard (N % 16 == 0) ---
    #pragma unroll
    for (int j = 0; j < 4; ++j) {
        const int colBase = n0 + wn * 64 + j * 16;
        if (colBase < N) {
            const v8f acc = (j == 0) ? acc0 : (j == 1) ? acc1 : (j == 2) ? acc2 : acc3;
            const int col = colBase + l15;
            #pragma unroll
            for (int r = 0; r < 8; ++r) {
                const int row = m0 + wm * 16 + r + 8 * half;
                float v = acc[r];
                if (bias) v += bias[col];
                if (Cacc) v += Cacc[(long long)b * Cbat + (long long)row * Crs + col];
                if (act == 1) v = tanhf(v);
                C[(long long)b * Cbat + (long long)row * Crs + col] = v;
            }
        }
    }
}

// ---------------------------------------------------------------------------
// GRU cell pointwise update for one timestep (PyTorch gate equations).
// ---------------------------------------------------------------------------
__global__ void gru_cell_kernel(const float* __restrict__ gi,
                                const float* __restrict__ gh,
                                float* __restrict__ h,
                                float* __restrict__ out, int t)
{
    const int idx = blockIdx.x * blockDim.x + threadIdx.x;
    if (idx >= BSZ * H2) return;
    const int b = idx / H2, j = idx % H2;

    const long long gib = ((long long)b * TDEC + t) * H3;
    const long long ghb = (long long)b * H3;

    const float ir = gi[gib + j],          hr = gh[ghb + j];
    const float iz = gi[gib + H2 + j],     hz = gh[ghb + H2 + j];
    const float in = gi[gib + 2 * H2 + j], hn = gh[ghb + 2 * H2 + j];

    const float r = 1.0f / (1.0f + expf(-(ir + hr)));
    const float z = 1.0f / (1.0f + expf(-(iz + hz)));
    const float n = tanhf(in + r * hn);

    const float hp = h[(long long)b * H2 + j];
    const float hnew = (1.0f - z) * n + z * hp;
    h[(long long)b * H2 + j] = hnew;
    out[((long long)b * TDEC + t) * H2 + j] = hnew;
}

// ---------------------------------------------------------------------------
// Masked softmax over encoder positions (axis=1 of s[b, s_enc, t_dec]).
// ---------------------------------------------------------------------------
__global__ void attn_softmax_kernel(const float* __restrict__ s,
                                    const int* __restrict__ mask,
                                    float* __restrict__ aw)
{
    const int idx = blockIdx.x * blockDim.x + threadIdx.x;
    if (idx >= BSZ * TDEC) return;
    const int b = idx / TDEC, t = idx % TDEC;

    const float* sb = s  + (long long)b * TENC * TDEC + t;
    float*       ab = aw + (long long)b * TENC * TDEC + t;
    const int m = mask[b];

    float mx = -3.0e38f;
    for (int se = 0; se < TENC; ++se) {
        if (se < m) mx = fmaxf(mx, sb[(long long)se * TDEC]);
    }
    float sum = 0.0f;
    float e[TENC];
    for (int se = 0; se < TENC; ++se) {
        const float v = (se < m) ? expf(sb[(long long)se * TDEC] - mx) : 0.0f;
        e[se] = v;
        sum += v;
    }
    const float inv = 1.0f / sum;
    for (int se = 0; se < TENC; ++se) {
        ab[(long long)se * TDEC] = e[se] * inv;
    }
}

// ---------------------------------------------------------------------------
// Host-side GEMM launcher.  Requires M % 32 == 0, N % 16 == 0.
// ---------------------------------------------------------------------------
static void launch_gemm(const float* A, const float* B, const float* bias,
                        const float* Cacc, float* C,
                        int M, int N, int K,
                        long long Ars, long long Acs,
                        long long Brs, long long Bcs, long long Crs,
                        long long Abat, long long Bbat, long long Cbat,
                        int batches, int act, hipStream_t stream)
{
    dim3 grid((unsigned)((N + BN - 1) / BN), (unsigned)(M / BM), (unsigned)batches);
    wmma_gemm_kernel<<<grid, 256, 0, stream>>>(
        A, B, bias, Cacc, C, M, N, K,
        Ars, Acs, Brs, Bcs, Crs, Abat, Bbat, Cbat, act);
}

extern "C" void kernel_launch(void* const* d_in, const int* in_sizes, int n_in,
                              void* d_out, int out_size, void* d_ws, size_t ws_size,
                              hipStream_t stream) {
    (void)in_sizes; (void)n_in; (void)out_size; (void)ws_size;

    const float* emb   = (const float*)d_in[0];   // [B,TDEC,GLOVE]
    const float* hs    = (const float*)d_in[1];   // [B,TENC,H2]
    const float* h_in  = (const float*)d_in[2];   // [2,B,H2]
    const int*   mask  = (const int*)  d_in[3];   // [B]
    const float* W_in  = (const float*)d_in[4];   // [H2,GLOVE]
    const float* b_in  = (const float*)d_in[5];
    const float* Wih0  = (const float*)d_in[6];   // [3H,H2]
    const float* Whh0  = (const float*)d_in[7];
    const float* bih0  = (const float*)d_in[8];
    const float* bhh0  = (const float*)d_in[9];
    const float* Wih1  = (const float*)d_in[10];
    const float* Whh1  = (const float*)d_in[11];
    const float* bih1  = (const float*)d_in[12];
    const float* bhh1  = (const float*)d_in[13];
    const float* W_cat = (const float*)d_in[14];  // [H2, 2H2]
    const float* b_cat = (const float*)d_in[15];
    const float* W_out = (const float*)d_in[16];  // [VOCAB,H2]
    const float* b_out = (const float*)d_in[17];

    // d_out layout: logits [BT,VOCAB] | h_new [2,B,H2] | aw [B,TENC,TDEC]
    float* logits = (float*)d_out;
    float* h_new  = logits + (long long)BT * VOCAB;
    float* aw     = h_new + 2LL * BSZ * H2;

    // workspace layout (fp32 elements)
    float* ws   = (float*)d_ws;
    float* x    = ws;                          // [BT, H2]
    float* gi   = x    + (long long)BT * H2;   // [BT, 3H] (reused per layer)
    float* out0 = gi   + (long long)BT * H3;   // [BT, H2]
    float* out1 = out0 + (long long)BT * H2;   // [BT, H2]
    float* gh   = out1 + (long long)BT * H2;   // [B, 3H]
    float* hcur = gh   + (long long)BSZ * H3;  // [2, B, H2]
    float* sbuf = hcur + 2LL * BSZ * H2;       // [B, TENC, TDEC]
    float* cbuf = sbuf + (long long)BSZ * TENC * TDEC; // [BT, H2]
    float* tbuf = cbuf + (long long)BT * H2;   // [BT, H2]

    // live hidden state
    hipMemcpyAsync(hcur, h_in, 2LL * BSZ * H2 * sizeof(float),
                   hipMemcpyDeviceToDevice, stream);

    // 1) input projection: x = emb @ W_in^T + b_in   (K=300: last tile generic)
    launch_gemm(emb, W_in, b_in, nullptr, x,
                BT, H2, GLOVE,
                /*Ars*/GLOVE, /*Acs*/1, /*Brs*/1, /*Bcs*/GLOVE, /*Crs*/H2,
                0, 0, 0, /*batches*/1, /*act*/0, stream);

    const int cellThreads = BSZ * H2;
    const int cellBlocks = (cellThreads + 255) / 256;

    // 2) GRU layer 0: hoisted input gates, then sequential recurrence
    launch_gemm(x, Wih0, bih0, nullptr, gi,
                BT, H3, H2, H2, 1, 1, H2, H3, 0, 0, 0, 1, 0, stream);
    for (int t = 0; t < TDEC; ++t) {
        launch_gemm(hcur, Whh0, bhh0, nullptr, gh,
                    BSZ, H3, H2, H2, 1, 1, H2, H3, 0, 0, 0, 1, 0, stream);
        gru_cell_kernel<<<cellBlocks, 256, 0, stream>>>(gi, gh, hcur, out0, t);
    }

    // 3) GRU layer 1
    launch_gemm(out0, Wih1, bih1, nullptr, gi,
                BT, H3, H2, H2, 1, 1, H2, H3, 0, 0, 0, 1, 0, stream);
    for (int t = 0; t < TDEC; ++t) {
        launch_gemm(hcur + (long long)BSZ * H2, Whh1, bhh1, nullptr, gh,
                    BSZ, H3, H2, H2, 1, 1, H2, H3, 0, 0, 0, 1, 0, stream);
        gru_cell_kernel<<<cellBlocks, 256, 0, stream>>>(
            gi, gh, hcur + (long long)BSZ * H2, out1, t);
    }

    // 4) attention scores: s[b,se,td] = sum_e hs[b,se,e] * out1[b,td,e]
    launch_gemm(hs, out1, nullptr, nullptr, sbuf,
                TENC, TDEC, H2,
                /*Ars*/H2, /*Acs*/1, /*Brs*/1, /*Bcs*/H2, /*Crs*/TDEC,
                /*Abat*/(long long)TENC * H2, /*Bbat*/(long long)TDEC * H2,
                /*Cbat*/(long long)TENC * TDEC, /*batches*/BSZ, 0, stream);

    // 5) masked softmax over encoder axis -> aw (directly into d_out slot)
    {
        const int n = BSZ * TDEC;
        attn_softmax_kernel<<<(n + 255) / 256, 256, 0, stream>>>(sbuf, mask, aw);
    }

    // 6) context: c[b,t,e] = sum_s aw[b,s,t] * hs[b,s,e]  (A is aw^T via strides)
    launch_gemm(aw, hs, nullptr, nullptr, cbuf,
                TDEC, H2, TENC,
                /*Ars*/1, /*Acs*/TDEC, /*Brs*/H2, /*Bcs*/1, /*Crs*/H2,
                /*Abat*/(long long)TENC * TDEC, /*Bbat*/(long long)TENC * H2,
                /*Cbat*/(long long)TDEC * H2, /*batches*/BSZ, 0, stream);

    // 7) cat-projection without materializing cat:
    //    tbuf = out1 @ Wcat[:, :H2]^T + b_cat ; tbuf = tanh(tbuf + c @ Wcat[:, H2:]^T)
    launch_gemm(out1, W_cat, b_cat, nullptr, tbuf,
                BT, H2, H2,
                H2, 1, /*Brs*/1, /*Bcs*/2 * H2, H2, 0, 0, 0, 1, /*act*/0, stream);
    launch_gemm(cbuf, W_cat + H2, nullptr, /*Cacc*/tbuf, tbuf,
                BT, H2, H2,
                H2, 1, /*Brs*/1, /*Bcs*/2 * H2, H2, 0, 0, 0, 1, /*act*/1, stream);

    // 8) vocab projection (dominant GEMM, 134 GFLOP): logits = tbuf @ W_out^T + b_out
    launch_gemm(tbuf, W_out, b_out, nullptr, logits,
                BT, VOCAB, H2,
                H2, 1, /*Brs*/1, /*Bcs*/H2, VOCAB, 0, 0, 0, 1, 0, stream);

    // 9) h_new
    hipMemcpyAsync(h_new, hcur, 2LL * BSZ * H2 * sizeof(float),
                   hipMemcpyDeviceToDevice, stream);
}
